// Generator_2937757631010
// MI455X (gfx1250) — matrix operand order repaired and checked by reading, not compiled, not run on hardware
//
#include <hip/hip_runtime.h>
#include <hip/hip_bf16.h>

// ---------------------------------------------------------------------------
// Problem constants (from the reference)
// ---------------------------------------------------------------------------
#define ATTR     4
#define N_BASIS  10
#define D_SH     9
#define MUL      64      // W1: [10,64]  W2: [64,128]  W3: [128,do]
#define MUL2     128

typedef __attribute__((ext_vector_type(16))) _Float16 v16h;
typedef __attribute__((ext_vector_type(8)))  _Float16 v8h;
typedef __attribute__((ext_vector_type(8)))  float    v8f;

// ---------------------------------------------------------------------------
// WMMA fragment index helpers (CDNA5 16x16x32 f16 layouts, CDNA5 ISA)
// A (16x32 MxK): lane L -> m = L&15, hi = L>>4; element j:
//   j in [0,8):  k = hi*8 + j          (contiguous run of 8)
//   j in [8,16): k = 16 + hi*8 + (j-8) (contiguous run of 8)
// B (32x16 KxN): mirrored (lane holds column n = L&15, same k pattern)
// C/D (16x16 f32): lane L -> n = L&15; element r -> m = r + 8*(L>>4)
// ---------------------------------------------------------------------------
__device__ __forceinline__ int frag_k(int j, int hi) {
    int v = j >> 1, p = j & 1;
    return ((v & 4) ? 16 : 0) + (hi << 3) + ((v & 3) << 1) + p;
}

// fast silu: v * sigmoid(v), sigmoid via v_exp_f32 + v_rcp_f32 (no IEEE div)
__device__ __forceinline__ float silu_f(float v) {
    float e = __builtin_amdgcn_exp2f(v * -1.442695040888963f);
    return v * __builtin_amdgcn_rcpf(1.0f + e);
}

__device__ __forceinline__ v8f wmma16(v16h a, v16h b, v8f c) {
    return __builtin_amdgcn_wmma_f32_16x16x32_f16(
        /*neg_a=*/false, a, /*neg_b=*/false, b,
        /*c_mod=*/(short)0, c, /*reuse_a=*/false, /*reuse_b=*/false);
}

// Load one fragment (A or B) from f16 LDS: two 8-half runs -> 2x ds_load_b128.
__device__ __forceinline__ v16h lds_a_frag(const _Float16* row, int hi) {
    v8h lo = *(const v8h*)(row + hi * 8);
    v8h hi8 = *(const v8h*)(row + 16 + hi * 8);
    return __builtin_shufflevector(lo, hi8, 0, 1, 2, 3, 4, 5, 6, 7,
                                   8, 9, 10, 11, 12, 13, 14, 15);
}

// ---------------------------------------------------------------------------
// Prepack a row-major f32 weight [K x Nn] into per-lane B-fragment order:
//   frag f = kt*NT + nt, then 32 lanes x 16 halves contiguous (512 halves/frag)
// ---------------------------------------------------------------------------
__global__ void prepack_b_kernel(const float* __restrict__ W,
                                 _Float16* __restrict__ out,
                                 int K, int Nn, int KT, int NT) {
    int idx = blockIdx.x * blockDim.x + threadIdx.x;
    int total = KT * NT * 512;
    if (idx >= total) return;
    int f    = idx >> 9;
    int r    = idx & 511;
    int lane = r >> 4;
    int j    = r & 15;
    int kt = f / NT, nt = f % NT;
    int hi = lane >> 4;
    int n  = nt * 16 + (lane & 15);
    int k  = kt * 32 + frag_k(j, hi);
    float v = (k < K && n < Nn) ? W[k * Nn + n] : 0.0f;
    out[idx] = (_Float16)v;
}

// ---------------------------------------------------------------------------
// Per-node prep: xa = x @ A ; acc = self-connection sc
// ---------------------------------------------------------------------------
template<int DI, int DO, int NB>
__global__ __launch_bounds__(128)
void node_prep_kernel(const float* __restrict__ x,
                      const float* __restrict__ attr,
                      const float* __restrict__ A,    // [DI, DO]
                      const float* __restrict__ Wsc,  // [DI*ATTR, DO]
                      float* __restrict__ xa,         // [N, DO]
                      float* __restrict__ acc,        // [N, DO]
                      int n_nodes) {
    __shared__ float xs[NB][DI];
    __shared__ float as[NB][ATTR];
    int n0 = blockIdx.x * NB;
    for (int i = threadIdx.x; i < NB * DI; i += blockDim.x) {
        int nb = i / DI, ii = i % DI;
        int n = n0 + nb;
        xs[nb][ii] = (n < n_nodes) ? x[(size_t)n * DI + ii] : 0.0f;
    }
    for (int i = threadIdx.x; i < NB * ATTR; i += blockDim.x) {
        int nb = i / ATTR, jj = i % ATTR;
        int n = n0 + nb;
        as[nb][jj] = (n < n_nodes) ? attr[(size_t)n * ATTR + jj] : 0.0f;
    }
    __syncthreads();
    for (int o = threadIdx.x; o < DO; o += blockDim.x) {
        float sxa[NB], ssc[NB];
#pragma unroll
        for (int nb = 0; nb < NB; ++nb) { sxa[nb] = 0.0f; ssc[nb] = 0.0f; }
        for (int i = 0; i < DI; ++i) {
            float a = A[i * DO + o];
            float w[ATTR];
#pragma unroll
            for (int j = 0; j < ATTR; ++j) w[j] = Wsc[(i * ATTR + j) * DO + o];
#pragma unroll
            for (int nb = 0; nb < NB; ++nb) {
                float xi = xs[nb][i];
                sxa[nb] += xi * a;
                float t = 0.0f;
#pragma unroll
                for (int j = 0; j < ATTR; ++j) t += as[nb][j] * w[j];
                ssc[nb] += xi * t;
            }
        }
#pragma unroll
        for (int nb = 0; nb < NB; ++nb) {
            int n = n0 + nb;
            if (n < n_nodes) {
                xa [(size_t)n * DO + o] = sxa[nb];
                acc[(size_t)n * DO + o] = ssc[nb];
            }
        }
    }
}

// ---------------------------------------------------------------------------
// Elementwise silu
// ---------------------------------------------------------------------------
__global__ void silu_kernel(const float* __restrict__ in, float* __restrict__ out,
                            int count) {
    int i = blockIdx.x * blockDim.x + threadIdx.x;
    for (; i < count; i += gridDim.x * blockDim.x) out[i] = silu_f(in[i]);
}

// ---------------------------------------------------------------------------
// Fused edge kernel: per 16-edge tile (one wave each):
//   gate = silu(silu(emb@W1)@W2)@W3         (WMMA, f16 in / f32 acc)
//   shb  = sh@B                             (1 WMMA per output tile)
//   msg  = gate * shb * xa[src] / sqrt(32)  (register product in C layout)
//   atomicAdd into acc[dst]
// w1/w2/bsh fragments staged block-wide in LDS (low-latency ds_load_b128 ->
// no long-range prefetch pressure); w3 streamed from global (L0-resident).
// ---------------------------------------------------------------------------
#define SCR_STRIDE 136   // halves; 128 cols + pad

template<int DO, int DT>   // DT = number of 16-wide output tiles
__global__ __launch_bounds__(128, 4)
void edge_kernel(const _Float16* __restrict__ w1p,   // 1x4  frags  (K=32 ,N=64)
                 const _Float16* __restrict__ w2p,   // 2x8  frags  (K=64 ,N=128)
                 const _Float16* __restrict__ w3p,   // 4xDT frags  (K=128,N=DO)
                 const _Float16* __restrict__ bshp,  // 1xDT frags  (K=32 ,N=DO)
                 const float* __restrict__ emb,      // [E, 10]
                 const float* __restrict__ sh,       // [E, 9]
                 const float* __restrict__ xa,       // [N, DO]
                 const int*   __restrict__ src,
                 const int*   __restrict__ dst,
                 float* __restrict__ acc,            // [N, DO]
                 int n_edges, int n_tiles) {
    // per-wave f16 LDS transpose scratch + input staging
    __shared__ _Float16 scratch_all[4][16 * SCR_STRIDE];
    __shared__ _Float16 estage_all[4][16 * 32];   // emb tile, zero-padded cols
    __shared__ _Float16 shstage_all[4][16 * 32];  // sh tile,  zero-padded cols
    // block-wide f16 weight fragments: w1 (4 frags) | w2 (16 frags) | bsh (DT)
    __shared__ _Float16 wlds[(4 + 16 + DT) * 512];
    const int W2OFF = 4 * 512;
    const int BSOFF = 20 * 512;

    const int lane = threadIdx.x & 31;
    const int wave = threadIdx.x >> 5;
    const int tid  = threadIdx.x;
    _Float16* scr = &scratch_all[wave][0];
    _Float16* est = &estage_all[wave][0];
    _Float16* shs = &shstage_all[wave][0];
    const int m  = lane & 15;   // A row / C column within tile
    const int hi = lane >> 4;
    const float inv_nn = 0.17677669529663687f;  // 1/sqrt(32)

    // stage small weight arrays into LDS (once per block)
    for (int i = tid * 8; i < 4 * 512; i += 128 * 8)
        *(v8h*)(wlds + i) = *(const v8h*)(w1p + i);
    for (int i = tid * 8; i < 16 * 512; i += 128 * 8)
        *(v8h*)(wlds + W2OFF + i) = *(const v8h*)(w2p + i);
    for (int i = tid * 8; i < DT * 512; i += 128 * 8)
        *(v8h*)(wlds + BSOFF + i) = *(const v8h*)(bshp + i);
    // zero-init staging pads (only cols < 10 / < 9 are rewritten per tile)
    for (int i = lane; i < 16 * 32; i += 32) { est[i] = (_Float16)0.0f; shs[i] = (_Float16)0.0f; }
    __syncthreads();

    int wg = blockIdx.x * 4 + wave;
    int ws = gridDim.x * 4;
    for (int t = wg; t < n_tiles; t += ws) {
        const int e0 = t * 16;
        const int nv = min(16, n_edges - e0);
        // ---- coalesced stage of emb/sh tiles (contiguous rows) -------------
        if (nv == 16) {
            const float* eb = emb + (size_t)e0 * N_BASIS;
#pragma unroll
            for (int it = 0; it < 5; ++it) {
                int i = lane + it * 32;
                est[(i / N_BASIS) * 32 + (i % N_BASIS)] = (_Float16)eb[i];
            }
            const float* sb = sh + (size_t)e0 * D_SH;
#pragma unroll
            for (int it = 0; it < 5; ++it) {
                int i = lane + it * 32;
                if (i < 16 * D_SH)
                    shs[(i / D_SH) * 32 + (i % D_SH)] = (_Float16)sb[i];
            }
        } else {
            const float* eb = emb + (size_t)e0 * N_BASIS;
            for (int i = lane; i < nv * N_BASIS; i += 32)
                est[(i / N_BASIS) * 32 + (i % N_BASIS)] = (_Float16)eb[i];
            const float* sb = sh + (size_t)e0 * D_SH;
            for (int i = lane; i < nv * D_SH; i += 32)
                shs[(i / D_SH) * 32 + (i % D_SH)] = (_Float16)sb[i];
        }
        v16h a0  = lds_a_frag(est + m * 32, hi);
        v16h ash = lds_a_frag(shs + m * 32, hi);
        // ---- GEMM1: h1 = silu(emb @ W1)  -> scratch cols 0..63 -------------
#pragma unroll
        for (int nt = 0; nt < 4; ++nt) {
            v16h b = __builtin_shufflevector(*(const v8h*)(wlds + nt * 512 + lane * 16), *(const v8h*)(wlds + nt * 512 + lane * 16 + 8), 0, 1, 2, 3, 4, 5, 6, 7, 8, 9, 10, 11, 12, 13, 14, 15);
            v8f c = {};
            c = wmma16(a0, b, c);
            int col = nt * 16 + m;
#pragma unroll
            for (int r = 0; r < 8; ++r)
                scr[(r + 8 * hi) * SCR_STRIDE + col] = (_Float16)silu_f(c[r]);
        }
        // ---- A frags for GEMM2 (K = 64): 2x ds_load_b128 each --------------
        v16h a2[2];
#pragma unroll
        for (int kt = 0; kt < 2; ++kt)
            a2[kt] = lds_a_frag(scr + m * SCR_STRIDE + kt * 32, hi);
        // ---- GEMM2: h2 = silu(h1 @ W2)  -> scratch cols 0..127 -------------
#pragma unroll
        for (int nt = 0; nt < 8; ++nt) {
            v8f c = {};
#pragma unroll
            for (int kt = 0; kt < 2; ++kt) {
                v16h b = __builtin_shufflevector(*(const v8h*)(wlds + W2OFF + (kt * 8 + nt) * 512 + lane * 16), *(const v8h*)(wlds + W2OFF + (kt * 8 + nt) * 512 + lane * 16 + 8), 0, 1, 2, 3, 4, 5, 6, 7, 8, 9, 10, 11, 12, 13, 14, 15);
                c = wmma16(a2[kt], b, c);
            }
            int col = nt * 16 + m;
#pragma unroll
            for (int r = 0; r < 8; ++r)
                scr[(r + 8 * hi) * SCR_STRIDE + col] = (_Float16)silu_f(c[r]);
        }
        // ---- A frags for GEMM3 (K = 128) -----------------------------------
        v16h a3[4];
#pragma unroll
        for (int kt = 0; kt < 4; ++kt)
            a3[kt] = lds_a_frag(scr + m * SCR_STRIDE + kt * 32, hi);
        // ---- src/dst for the 8 edge-rows this lane owns (clamped loads) ----
        int ssrc[8], sdst[8];
#pragma unroll
        for (int r = 0; r < 8; ++r) {
            int ee  = e0 + r + 8 * hi;
            int eec = (ee < n_edges) ? ee : (n_edges - 1);
            ssrc[r] = src[eec];
            int d   = dst[eec];
            sdst[r] = (ee < n_edges) ? d : -1;
        }
        // ---- GEMM3 (gate) + shb WMMA per output tile, combine, scatter -----
#pragma unroll 1
        for (int nt = 0; nt < DT; ++nt) {
            v8f g = {};
#pragma unroll
            for (int kt = 0; kt < 4; ++kt) {
                v16h b = *(const v16h*)(w3p + (size_t)(kt * DT + nt) * 512 + lane * 16);
                g = wmma16(a3[kt], b, g);
            }
            v16h bs = __builtin_shufflevector(*(const v8h*)(wlds + BSOFF + nt * 512 + lane * 16), *(const v8h*)(wlds + BSOFF + nt * 512 + lane * 16 + 8), 0, 1, 2, 3, 4, 5, 6, 7, 8, 9, 10, 11, 12, 13, 14, 15);
            v8f s = {};
            s = wmma16(ash, bs, s);
            int o = nt * 16 + m;
            if (o < DO) {
#pragma unroll
                for (int r = 0; r < 8; ++r) {
                    int d = sdst[r];
                    if (d >= 0) {
                        float val = g[r] * s[r] * xa[(size_t)ssrc[r] * DO + o] * inv_nn;
                        atomicAdd(&acc[(size_t)d * DO + o], val);
                    }
                }
            }
        }
    }
}

// ---------------------------------------------------------------------------
// Host launcher
// ---------------------------------------------------------------------------
static inline size_t align256(size_t x) { return (x + 255) & ~(size_t)255; }

extern "C" void kernel_launch(void* const* d_in, const int* in_sizes, int n_in,
                              void* d_out, int out_size, void* d_ws, size_t ws_size,
                              hipStream_t stream) {
    (void)n_in; (void)out_size; (void)ws_size;
    const float* x    = (const float*)d_in[0];
    const float* attr = (const float*)d_in[1];
    const float* sh   = (const float*)d_in[2];
    const float* emb  = (const float*)d_in[3];
    const float* A[3]   = {(const float*)d_in[4],  (const float*)d_in[10], (const float*)d_in[16]};
    const float* B[3]   = {(const float*)d_in[5],  (const float*)d_in[11], (const float*)d_in[17]};
    const float* W1[3]  = {(const float*)d_in[6],  (const float*)d_in[12], (const float*)d_in[18]};
    const float* W2[3]  = {(const float*)d_in[7],  (const float*)d_in[13], (const float*)d_in[19]};
    const float* W3[3]  = {(const float*)d_in[8],  (const float*)d_in[14], (const float*)d_in[20]};
    const float* Wsc[3] = {(const float*)d_in[9],  (const float*)d_in[15], (const float*)d_in[21]};
    const int* src = (const int*)d_in[22];
    const int* dst = (const int*)d_in[23];

    const int N = in_sizes[0] / 64;          // 20000
    const int E = in_sizes[2] / D_SH;        // 640000
    const int DHID = 144;
    const int DT_H = DHID / 16;              // 9
    const int n_tiles = (E + 15) / 16;

    // ---- workspace layout --------------------------------------------------
    char* p = (char*)d_ws;
    const size_t szH = align256((size_t)N * DHID * sizeof(float));
    float* hA   = (float*)p; p += szH;
    float* hB   = (float*)p; p += szH;
    float* xab  = (float*)p; p += szH;
    float* accb = (float*)p; p += szH;
    _Float16* w1p[3]; _Float16* w2p[3]; _Float16* w3p[3]; _Float16* bshp[3];
    const int DOs[3] = {DHID, DHID, 3};
    const int DTs[3] = {DT_H, DT_H, 1};
    for (int l = 0; l < 3; ++l) {
        w1p[l]  = (_Float16*)p; p += align256((size_t)1 * 4      * 512 * 2);
        w2p[l]  = (_Float16*)p; p += align256((size_t)2 * 8      * 512 * 2);
        w3p[l]  = (_Float16*)p; p += align256((size_t)4 * DTs[l] * 512 * 2);
        bshp[l] = (_Float16*)p; p += align256((size_t)1 * DTs[l] * 512 * 2);
    }

    // ---- prepack all weights into WMMA B-fragment order --------------------
    auto pack = [&](const float* W, _Float16* out, int K, int Nn, int KT, int NT) {
        int total = KT * NT * 512;
        prepack_b_kernel<<<(total + 255) / 256, 256, 0, stream>>>(W, out, K, Nn, KT, NT);
    };
    for (int l = 0; l < 3; ++l) {
        pack(W1[l], w1p[l],  N_BASIS, MUL,    1, 4);
        pack(W2[l], w2p[l],  MUL,     MUL2,   2, 8);
        pack(W3[l], w3p[l],  MUL2,    DOs[l], 4, DTs[l]);
        pack(B[l],  bshp[l], D_SH,    DOs[l], 1, DTs[l]);
    }

    const int npb0 = (N + 7) / 8;
    const int egrid = 2000;
    const int scount = N * DHID;

    // ---- layer 0: x(64) -> hA(144) -----------------------------------------
    node_prep_kernel<64, 144, 8><<<npb0, 128, 0, stream>>>(
        x, attr, A[0], Wsc[0], xab, accb, N);
    edge_kernel<144, 9><<<egrid, 128, 0, stream>>>(
        w1p[0], w2p[0], w3p[0], bshp[0], emb, sh, xab, src, dst, accb, E, n_tiles);
    silu_kernel<<<(scount + 255) / 256, 256, 0, stream>>>(accb, hA, scount);

    // ---- layer 1: hA(144) -> hB(144) ---------------------------------------
    node_prep_kernel<144, 144, 8><<<npb0, 128, 0, stream>>>(
        hA, attr, A[1], Wsc[1], xab, accb, N);
    edge_kernel<144, 9><<<egrid, 128, 0, stream>>>(
        w1p[1], w2p[1], w3p[1], bshp[1], emb, sh, xab, src, dst, accb, E, n_tiles);
    silu_kernel<<<(scount + 255) / 256, 256, 0, stream>>>(accb, hB, scount);

    // ---- layer 2: hB(144) -> d_out(3), accumulate straight into d_out ------
    float* out = (float*)d_out;
    node_prep_kernel<144, 3, 8><<<npb0, 128, 0, stream>>>(
        hB, attr, A[2], Wsc[2], xab, out, N);
    edge_kernel<3, 1><<<egrid, 128, 0, stream>>>(
        w1p[2], w2p[2], w3p[2], bshp[2], emb, sh, xab, src, dst, out, E, n_tiles);
}
